// EdgeEmbedding_67353677136593
// MI455X (gfx1250) — matrix operand order repaired and checked by reading, not compiled, not run on hardware
//
#include <hip/hip_runtime.h>
#include <hip/hip_bf16.h>

typedef __attribute__((ext_vector_type(16))) _Float16 v16h;
typedef __attribute__((ext_vector_type(8)))  float    v8f;

#define EMB   64
#define RAD   32
#define NSPEC 100
#define CUTOFF_F 5.0f

union HFrag {
    v16h     v;
    _Float16 h[16];
    uint4    u[2];
};

__device__ __forceinline__ float bpermf(int srcLane, float val) {
    return __int_as_float(__builtin_amdgcn_ds_bpermute(srcLane << 2, __float_as_int(val)));
}

__global__ __launch_bounds__(256, 1)
void edge_embed_wmma(const int*   __restrict__ Z,
                     const int*   __restrict__ nidx,
                     const float* __restrict__ nvec,
                     const float* __restrict__ zemb,
                     const float* __restrict__ zw,
                     const float* __restrict__ rw,
                     const float* __restrict__ rb,
                     const float* __restrict__ cen,
                     const float* __restrict__ bet,
                     float*       __restrict__ out,
                     int E, int ntiles)
{
    __shared__ __align__(16) _Float16 sEmb[NSPEC * EMB];      // 100*64  f16
    __shared__ __align__(16) _Float16 sWz[EMB * 2 * EMB];     // 64*128  f16 (row n, col k)
    __shared__ __align__(16) _Float16 sWr[3 * EMB * RAD];     // 192*32  f16 (row n, col k)
    __shared__ float sRb[3 * EMB];
    __shared__ float sCen[RAD];
    __shared__ float sBet[RAD];

    const int tid = threadIdx.x;
    for (int i = tid; i < NSPEC * EMB;   i += 256) sEmb[i] = (_Float16)zemb[i];
    for (int i = tid; i < EMB * 2 * EMB; i += 256) sWz[i]  = (_Float16)zw[i];
    for (int i = tid; i < 3 * EMB * RAD; i += 256) sWr[i]  = (_Float16)rw[i];
    for (int i = tid; i < 3 * EMB;       i += 256) sRb[i]  = rb[i];
    if (tid < RAD) { sCen[tid] = cen[tid]; sBet[tid] = bet[tid]; }
    __syncthreads();

    const int wave = tid >> 5;
    const int lane = tid & 31;
    const int tile = blockIdx.x * 8 + wave;
    if (tile >= ntiles) return;                    // wave-uniform branch: EXEC stays full

    const int  m    = lane & 15;                   // edge-in-tile owned by this lane
    const int  half = lane >> 4;
    const long base = (long)tile * 16;
    long e = base + m; if (e > (long)(E - 1)) e = (long)(E - 1);

    // ---- per-edge scalars --------------------------------------------------
    const int ia = nidx[2 * e];
    const int ib = nidx[2 * e + 1];
    const int Zi = Z[ia];
    const int Zj = Z[ib];
    const float vx = nvec[3 * e], vy = nvec[3 * e + 1], vz = nvec[3 * e + 2];
    const float r   = sqrtf(vx * vx + vy * vy + vz * vz);
    const float inv = 1.0f / r;
    const float x = vx * inv, y = vy * inv, z = vz * inv;
    const float er  = __expf(-r);
    const float env = (r < CUTOFF_F)
                        ? 0.5f * (__cosf(0.62831853071795865f * r) + 1.0f)   // pi/5
                        : 0.0f;

    // ---- RBF expansion as WMMA A-fragment (16 edges x K=32, f16) ----------
    // 16-bit A layout: lanes 0-15 hold K 0-7 (v0-3) and 16-23 (v4-7);
    //                  lanes 16-31 hold K 8-15 and 24-31.
    HFrag aExp;
    #pragma unroll
    for (int kk = 0; kk < 8; ++kk) {
        const int k0 = half * 8 + kk;
        const int k1 = 16 + half * 8 + kk;
        const float d0 = er - sCen[k0];
        const float d1 = er - sCen[k1];
        aExp.h[kk]     = (_Float16)__expf(-sBet[k0] * d0 * d0);
        aExp.h[8 + kk] = (_Float16)__expf(-sBet[k1] * d1 * d1);
    }

    // ---- Hz A-fragments: concat(emb[Zi], emb[Zj]) = 16 x 128, 4 K-frags ---
    HFrag aHz[4];
    #pragma unroll
    for (int t = 0; t < 4; ++t) {
        const int sp = (t < 2) ? Zi : Zj;
        const _Float16* rowp = &sEmb[sp * EMB + (t & 1) * 32];
        aHz[t].u[0] = *(const uint4*)(rowp + half * 8);        // K 0-7  of frag
        aHz[t].u[1] = *(const uint4*)(rowp + 16 + half * 8);   // K 16-23 of frag
    }

    // ---- GEMM1: Hz(16x128) @ z_map_w^T(128x64) via 16 WMMAs ---------------
    // B 16-bit layout: lanes 0-15 = col n, K 0-7 and 16-23; lanes 16-31 = col n, K 8-15 and 24-31.
    v8f acc1[4];
    #pragma unroll
    for (int nt = 0; nt < 4; ++nt) {
        v8f c = {0.f, 0.f, 0.f, 0.f, 0.f, 0.f, 0.f, 0.f};
        #pragma unroll
        for (int t = 0; t < 4; ++t) {
            HFrag b;
            const _Float16* bp = &sWz[(nt * 16 + m) * 128 + t * 32 + half * 8];
            b.u[0] = *(const uint4*)bp;
            b.u[1] = *(const uint4*)(bp + 16);
            c = __builtin_amdgcn_wmma_f32_16x16x32_f16(false, aHz[t].v,
                                                       false, b.v,
                                                       (short)0, c, false, false);
        }
        acc1[nt] = c;                                           // hz tile, cols nt*16..+15
    }

    // ---- broadcast per-row scalars to C/D row owners (rows half*8+v) ------
    float envR[8], xR[8], yR[8], zR[8];
    #pragma unroll
    for (int v = 0; v < 8; ++v) {
        const int src = half * 8 + v;
        envR[v] = bpermf(src, env);
        xR[v]   = bpermf(src, x);
        yR[v]   = bpermf(src, y);
        zR[v]   = bpermf(src, z);
    }

    const long E576 = (long)E * 576;

    // ---- GEMM2 tiles fused with bias/env/hz and 3x3 expansion -------------
    #pragma unroll
    for (int j = 0; j < 12; ++j) {
        HFrag b;
        const _Float16* bp = &sWr[(j * 16 + m) * RAD + half * 8];
        b.u[0] = *(const uint4*)bp;
        b.u[1] = *(const uint4*)(bp + 16);
        v8f c2 = {0.f, 0.f, 0.f, 0.f, 0.f, 0.f, 0.f, 0.f};
        c2 = __builtin_amdgcn_wmma_f32_16x16x32_f16(false, aExp.v,
                                                    false, b.v,
                                                    (short)0, c2, false, false);

        const float bias = sRb[j * 16 + m];
        const int   g    = j >> 2;                 // 0=I, 1=A, 2=S
        const int   q    = (j & 3) * 16 + m;       // embedding channel 0..63
        float* outg      = out + (long)g * E576;
        const v8f hz     = acc1[j & 3];            // same lane/slot as c2 -> lane-local fuse

        #pragma unroll
        for (int v = 0; v < 8; ++v) {
            const long eo = base + half * 8 + v;
            if (eo < E) {
                const float cv = (c2[v] + bias) * envR[v] * hz[v];
                float* p = outg + eo * 576 + (long)q * 9;
                if (g == 0) {
                    p[0] = cv;  p[1] = 0.f; p[2] = 0.f;
                    p[3] = 0.f; p[4] = cv;  p[5] = 0.f;
                    p[6] = 0.f; p[7] = 0.f; p[8] = cv;
                } else if (g == 1) {
                    const float ax = xR[v], ay = yR[v], az = zR[v];
                    p[0] = 0.f;       p[1] = -cv * az; p[2] =  cv * ay;
                    p[3] =  cv * az;  p[4] = 0.f;      p[5] = -cv * ax;
                    p[6] = -cv * ay;  p[7] =  cv * ax; p[8] = 0.f;
                } else {
                    const float ax = xR[v], ay = yR[v], az = zR[v];
                    const float tt  = (ax * ax + ay * ay + az * az) * (1.0f / 3.0f);
                    const float s01 = cv * ax * ay;
                    const float s02 = cv * ax * az;
                    const float s12 = cv * ay * az;
                    p[0] = cv * (ax * ax - tt); p[1] = s01; p[2] = s02;
                    p[3] = s01; p[4] = cv * (ay * ay - tt); p[5] = s12;
                    p[6] = s02; p[7] = s12; p[8] = cv * (az * az - tt);
                }
            }
        }
    }
}

extern "C" void kernel_launch(void* const* d_in, const int* in_sizes, int n_in,
                              void* d_out, int out_size, void* d_ws, size_t ws_size,
                              hipStream_t stream) {
    (void)n_in; (void)out_size; (void)d_ws; (void)ws_size;
    const int E      = in_sizes[1] / 2;              // neighbour_index is (E,2)
    const int ntiles = (E + 15) / 16;                // 16 edges per wave tile
    const int blocks = (ntiles + 7) / 8;             // 8 waves (256 threads) per block

    edge_embed_wmma<<<blocks, 256, 0, stream>>>(
        (const int*)  d_in[0],   // Z
        (const int*)  d_in[1],   // neighbour_index
        (const float*)d_in[2],   // neighbour_vectors
        (const float*)d_in[3],   // z_embed
        (const float*)d_in[4],   // z_map_w
        (const float*)d_in[5],   // r_map_w
        (const float*)d_in[6],   // r_map_b
        (const float*)d_in[7],   // rbf_centers
        (const float*)d_in[8],   // rbf_betas
        (float*)d_out, E, ntiles);
}
